// GAT_mask_37056977830624
// MI455X (gfx1250) — matrix operand order repaired and checked by reading, not compiled, not run on hardware
//
#include <hip/hip_runtime.h>
#include <hip/hip_bf16.h>

// ---------------------------------------------------------------------------
// GAT (2 layers, heads=1) + BatchNorm/ELU + Linear head for gfx1250 (MI455X).
//
// Perf model: GEMMs (~7.5 GFLOP total) run on v_wmma_f32_16x16x32_f16 (K=128
// fully unrolled -> 4 WMMAs/tile, A and B^T fragments both as contiguous 16B
// loads) and are negligible; the dominant cost is the per-edge gather +
// atomic-scatter of 128-float rows (~3.5 GB total -> ~150us at 23.3 TB/s),
// done with coalesced float4 gathers + global_atomic_add_f32. Softmax max/sum
// are two light edge passes with float atomics; BN stats via LDS reduction.
// ---------------------------------------------------------------------------

typedef __attribute__((ext_vector_type(16))) _Float16 v16h;
typedef __attribute__((ext_vector_type(8)))  _Float16 v8h;
typedef __attribute__((ext_vector_type(8)))  float    v8f;

#define GN        100000
#define GE        1600000
#define GEN       (GE + GN)     // edges incl. self loops
#define HIDD      128
#define KDIM      128           // K of every GEMM in this net
#define NCLS      40
#define NCLS_PAD  48
#define NEG_SLOPE 0.2f
#define BN_EPS    1e-5f

// ---------------------------------------------------------------- utilities
__global__ void fill_f32(float* __restrict__ p, float v, int n) {
  int i = blockIdx.x * blockDim.x + threadIdx.x;
  int stride = gridDim.x * blockDim.x;
  for (; i < n; i += stride) p[i] = v;
}

// f32 -> f16 elementwise (same layout)
__global__ void convert_f16(const float* __restrict__ src,
                            _Float16* __restrict__ dst, long long total) {
  long long i = (long long)blockIdx.x * blockDim.x + threadIdx.x;
  long long stride = (long long)gridDim.x * blockDim.x;
  for (; i < total; i += stride) dst[i] = (_Float16)src[i];
}

// W[K x scols] f32 row-major  ->  Wt[dcols x K] f16 (transposed, zero-padded
// columns scols..dcols-1). Gives contiguous per-column K runs for B fragments.
__global__ void convert_transpose_f16(const float* __restrict__ src,
                                      _Float16* __restrict__ dst,
                                      int scols, int dcols) {
  int total = dcols * KDIM;
  int i = blockIdx.x * blockDim.x + threadIdx.x;
  int stride = gridDim.x * blockDim.x;
  for (; i < total; i += stride) {
    int c = i / KDIM;            // output row = column of W
    int k = i % KDIM;
    dst[i] = (c < scols) ? (_Float16)src[(long long)k * scols + c]
                         : (_Float16)0.0f;
  }
}

// ------------------------------------------------------------- WMMA GEMM
// C[M x Nc] = A[M x 128] * B[128 x Nc];  A f16 row-major, Bt = B^T f16
// (Nc x 128, row-major), C f32. One wave per 16x16 tile; 4 unrolled
// 16x16x32 WMMA steps. M, Nc multiples of 16.
__global__ __launch_bounds__(32) void wmma_gemm_f16f32(
    const _Float16* __restrict__ A, const _Float16* __restrict__ Bt,
    float* __restrict__ C, int Nc) {
  const int tileM = blockIdx.x * 16;
  const int tileN = blockIdx.y * 16;
  const int lane  = threadIdx.x;        // 0..31 (wave32)
  const int half  = lane >> 4;          // 0 | 1
  const int l16   = lane & 15;

  const _Float16* Arow = A  + (long long)(tileM + l16) * KDIM;  // lane's A row
  const _Float16* Bcol = Bt + (long long)(tileN + l16) * KDIM;  // lane's B col

  v8f acc = {};
  #pragma unroll
  for (int k0 = 0; k0 < KDIM; k0 += 32) {
    // A fragment: lanes 0-15 (rows 0-15) hold K=k0..k0+7 in v0-3 and
    // k0+16..k0+23 in v4-7; lanes 16-31 the +8 offsets. Two 16B loads.
    const _Float16* Ap = Arow + k0 + half * 8;
    v8h a_lo = *(const v8h*)(Ap);
    v8h a_hi = *(const v8h*)(Ap + 16);
    v16h afrag;
    #pragma unroll
    for (int j = 0; j < 8; ++j) { afrag[j] = a_lo[j]; afrag[j + 8] = a_hi[j]; }

    // B fragment: lane holds one column; same K layout as A (lanes 0-15
    // K=k0..k0+7, k0+16..k0+23; lanes 16-31 the +8 offsets). Two 16B loads.
    const _Float16* Bp = Bcol + k0 + half * 8;
    v8h b_lo = *(const v8h*)(Bp);
    v8h b_hi = *(const v8h*)(Bp + 16);
    v16h bfrag;
    #pragma unroll
    for (int j = 0; j < 8; ++j) { bfrag[j] = b_lo[j]; bfrag[j + 8] = b_hi[j]; }

    acc = __builtin_amdgcn_wmma_f32_16x16x32_f16(
        /*neg_a=*/false, afrag, /*neg_b=*/false, bfrag,
        /*c_mod=*/(short)0, acc, /*reuse_a=*/false, /*reuse_b=*/false);
  }
  // C/D layout: vgpr i, lane -> row = i + 8*half, col = l16
  #pragma unroll
  for (int i = 0; i < 8; ++i)
    C[(long long)(tileM + i + 8 * half) * Nc + (tileN + l16)] = acc[i];
}

// ------------------------------------------------- per-node attention scores
// One wave per node: e_src[n] = h[n].a_s, e_dst[n] = h[n].a_d  (float4 reads)
__global__ __launch_bounds__(256) void node_scores(
    const float* __restrict__ h, const float* __restrict__ a_s,
    const float* __restrict__ a_d, float* __restrict__ es,
    float* __restrict__ ed, int n) {
  int wid  = (blockIdx.x * blockDim.x + threadIdx.x) >> 5;
  int lane = threadIdx.x & 31;
  if (wid >= n) return;
  const float4 v  = ((const float4*)(h + (long long)wid * HIDD))[lane];
  const float4 as = ((const float4*)a_s)[lane];
  const float4 ad = ((const float4*)a_d)[lane];
  float s1 = v.x * as.x + v.y * as.y + v.z * as.z + v.w * as.w;
  float s2 = v.x * ad.x + v.y * ad.y + v.z * ad.z + v.w * ad.w;
  #pragma unroll
  for (int off = 16; off > 0; off >>= 1) {
    s1 += __shfl_xor(s1, off);
    s2 += __shfl_xor(s2, off);
  }
  if (lane == 0) { es[wid] = s1; ed[wid] = s2; }
}

// ----------------------------------------------------------- edge helpers
__device__ __forceinline__ float edge_score(const int* __restrict__ ei,
                                            const float* __restrict__ es,
                                            const float* __restrict__ ed,
                                            int i, int& s, int& d) {
  if (i < GE) { s = ei[i]; d = ei[GE + i]; }
  else        { s = d = i - GE; }   // self loop
  float e = es[s] + ed[d];
  return (e > 0.f) ? e : NEG_SLOPE * e;
}

__device__ __forceinline__ void atomicMaxF(float* addr, float val) {
  // monotone-in-bits trick; valid for non-NaN, addr initialized to -inf
  if (val >= 0.f) atomicMax((int*)addr, __float_as_int(val));
  else            atomicMin((unsigned int*)addr, __float_as_uint(val));
}

// pass A: segment max of edge scores per dst
__global__ void edge_max(const int* __restrict__ ei,
                         const float* __restrict__ es,
                         const float* __restrict__ ed,
                         float* __restrict__ mmax) {
  int i = blockIdx.x * blockDim.x + threadIdx.x;
  int stride = gridDim.x * blockDim.x;
  for (; i < GEN; i += stride) {
    int s, d;
    float e = edge_score(ei, es, ed, i, s, d);
    atomicMaxF(&mmax[d], e);
  }
}

// pass B: segment sum of exp(e - m[dst])
__global__ void edge_denom(const int* __restrict__ ei,
                           const float* __restrict__ es,
                           const float* __restrict__ ed,
                           const float* __restrict__ mmax,
                           float* __restrict__ denom) {
  int i = blockIdx.x * blockDim.x + threadIdx.x;
  int stride = gridDim.x * blockDim.x;
  for (; i < GEN; i += stride) {
    int s, d;
    float e = edge_score(ei, es, ed, i, s, d);
    atomicAdd(&denom[d], __expf(e - mmax[d]));
  }
}

// pass C: agg[dst] += alpha * h[src]
// One wave per edge: lane owns 4 contiguous feats (coalesced float4 gather,
// 4 global_atomic_add_f32 per lane). This pass is the HBM-roof bound one.
__global__ __launch_bounds__(256) void edge_aggregate(
    const int* __restrict__ ei, const float* __restrict__ es,
    const float* __restrict__ ed, const float* __restrict__ mmax,
    const float* __restrict__ denom, const float* __restrict__ h,
    float* __restrict__ agg) {
  long long wid = (long long)(blockIdx.x * blockDim.x + threadIdx.x) >> 5;
  int lane = threadIdx.x & 31;
  if (wid >= GEN) return;
  int i = (int)wid;
  int s, d;
  float e = edge_score(ei, es, ed, i, s, d);          // broadcast loads
  float alpha = __expf(e - mmax[d]) / denom[d];
  const float4 v = ((const float4*)(h + (long long)s * HIDD))[lane];
  float* ad = agg + (long long)d * HIDD + lane * 4;
  atomicAdd(ad + 0, alpha * v.x);
  atomicAdd(ad + 1, alpha * v.y);
  atomicAdd(ad + 2, alpha * v.z);
  atomicAdd(ad + 3, alpha * v.w);
}

// agg += bias (per feature)
__global__ void add_bias(float* __restrict__ agg, const float* __restrict__ b) {
  long long total = (long long)GN * HIDD;
  long long i = (long long)blockIdx.x * blockDim.x + threadIdx.x;
  long long stride = (long long)gridDim.x * blockDim.x;
  for (; i < total; i += stride) agg[i] += b[i % HIDD];
}

// ------------------------------------------------------------- BatchNorm
// per-feature sum / sumsq via LDS block reduction then 128-wide atomics
__global__ __launch_bounds__(256) void bn_stats(const float* __restrict__ z,
                                                float* __restrict__ stats) {
  __shared__ float s1[256], s2[256];
  int feat = threadIdx.x & (HIDD - 1);
  int sub  = threadIdx.x >> 7;          // 0 | 1
  float a = 0.f, b = 0.f;
  for (int row = blockIdx.x * 2 + sub; row < GN; row += gridDim.x * 2) {
    float v = z[(long long)row * HIDD + feat];
    a += v;
    b += v * v;
  }
  s1[threadIdx.x] = a;
  s2[threadIdx.x] = b;
  __syncthreads();
  if (sub == 0) {
    atomicAdd(&stats[feat],        s1[feat] + s1[feat + HIDD]);
    atomicAdd(&stats[HIDD + feat], s2[feat] + s2[feat + HIDD]);
  }
}

// zh = (z-mu)/sqrt(var+eps)*gamma+beta ; elu ; emit f16 for the next GEMM
__global__ void bn_elu_f16(const float* __restrict__ z,
                           const float* __restrict__ stats,
                           const float* __restrict__ gamma,
                           const float* __restrict__ beta,
                           _Float16* __restrict__ out16) {
  long long total = (long long)GN * HIDD;
  long long i = (long long)blockIdx.x * blockDim.x + threadIdx.x;
  long long stride = (long long)gridDim.x * blockDim.x;
  const float invN = 1.0f / (float)GN;
  for (; i < total; i += stride) {
    int f = (int)(i % HIDD);
    float mu  = stats[f] * invN;
    float var = stats[HIDD + f] * invN - mu * mu;
    float zh  = (z[i] - mu) * rsqrtf(var + BN_EPS) * gamma[f] + beta[f];
    float y   = (zh > 0.f) ? zh : (__expf(zh) - 1.0f);   // ELU(alpha=1)
    out16[i] = (_Float16)y;
  }
}

// head epilogue: out[n,40] = hpad[n,48] + bout
__global__ void out_bias(const float* __restrict__ hpad,
                         const float* __restrict__ bout,
                         float* __restrict__ out) {
  long long total = (long long)GN * NCLS;
  long long i = (long long)blockIdx.x * blockDim.x + threadIdx.x;
  long long stride = (long long)gridDim.x * blockDim.x;
  for (; i < total; i += stride) {
    int c = (int)(i % NCLS);
    long long r = i / NCLS;
    out[i] = hpad[r * NCLS_PAD + c] + bout[c];
  }
}

// ---------------------------------------------------------------------------
extern "C" void kernel_launch(void* const* d_in, const int* in_sizes, int n_in,
                              void* d_out, int out_size, void* d_ws, size_t ws_size,
                              hipStream_t stream) {
  const float* x      = (const float*)d_in[0];
  const int*   ei     = (const int*)d_in[1];     // [2,E] flat: src then dst
  const float* W1     = (const float*)d_in[2];
  const float* a_src1 = (const float*)d_in[3];
  const float* a_dst1 = (const float*)d_in[4];
  const float* b1     = (const float*)d_in[5];
  const float* W2     = (const float*)d_in[6];
  const float* a_src2 = (const float*)d_in[7];
  const float* a_dst2 = (const float*)d_in[8];
  const float* b2     = (const float*)d_in[9];
  const float* gamma  = (const float*)d_in[10];
  const float* beta   = (const float*)d_in[11];
  const float* Wout   = (const float*)d_in[12];
  const float* bout   = (const float*)d_in[13];
  float* out = (float*)d_out;

  // ---- workspace carving (256B aligned) --------------------------------
  const long long NB = (long long)GN * HIDD;
  char* ws = (char*)d_ws;
  size_t off = 0;
  auto carve = [&](size_t bytes) {
    void* p = ws + off;
    off = (off + bytes + 255) & ~(size_t)255;
    return p;
  };
  _Float16* hin16 = (_Float16*)carve(NB * sizeof(_Float16));          // GEMM input
  _Float16* wt16  = (_Float16*)carve((size_t)HIDD * KDIM * sizeof(_Float16));
  float*    h     = (float*)carve(NB * sizeof(float));                // GEMM output
  float*    agg   = (float*)carve(NB * sizeof(float));                // aggregated
  float*    es    = (float*)carve((size_t)GN * sizeof(float));
  float*    ed    = (float*)carve((size_t)GN * sizeof(float));
  float*    mmax  = (float*)carve((size_t)GN * sizeof(float));
  float*    denom = (float*)carve((size_t)GN * sizeof(float));
  float*    stats = (float*)carve(2 * HIDD * sizeof(float));
  (void)ws_size; (void)in_sizes; (void)n_in; (void)out_size;

  const int TB = 256;
  const int gridElem = 4096;                       // grid-stride elementwise
  const int gridEdge = (GEN + TB - 1) / TB;        // thread per edge
  const int gridWaveEdge = (int)(((long long)GEN * 32 + TB - 1) / TB);
  const int gridWaveNode = (GN * 32 + TB - 1) / TB;
  const dim3 gemmGridH(GN / 16, HIDD / 16);        // 6250 x 8
  const dim3 gemmGridO(GN / 16, NCLS_PAD / 16);    // 6250 x 3
  const float NEG_INF = -__builtin_huge_valf();

  // x -> f16 once
  convert_f16<<<gridElem, TB, 0, stream>>>(x, hin16, NB);

  const float* Ws[2]   = {W1, W2};
  const float* a_ss[2] = {a_src1, a_src2};
  const float* a_ds[2] = {a_dst1, a_dst2};
  const float* bs[2]   = {b1, b2};

  for (int layer = 0; layer < 2; ++layer) {
    // h = hin16 @ W (WMMA, f16 in / f32 accum); W staged as W^T f16
    convert_transpose_f16<<<64, TB, 0, stream>>>(Ws[layer], wt16, HIDD, HIDD);
    wmma_gemm_f16f32<<<gemmGridH, 32, 0, stream>>>(hin16, wt16, h, HIDD);

    // attention scores per node
    node_scores<<<gridWaveNode, TB, 0, stream>>>(h, a_ss[layer], a_ds[layer], es, ed, GN);

    // segment softmax over incoming edges + weighted aggregation
    fill_f32<<<gridElem, TB, 0, stream>>>(mmax, NEG_INF, GN);
    fill_f32<<<gridElem, TB, 0, stream>>>(denom, 0.f, GN);
    edge_max<<<gridEdge, TB, 0, stream>>>(ei, es, ed, mmax);
    edge_denom<<<gridEdge, TB, 0, stream>>>(ei, es, ed, mmax, denom);
    fill_f32<<<gridElem, TB, 0, stream>>>(agg, 0.f, (int)NB);
    edge_aggregate<<<gridWaveEdge, TB, 0, stream>>>(ei, es, ed, mmax, denom, h, agg);
    add_bias<<<gridElem, TB, 0, stream>>>(agg, bs[layer]);

    // BatchNorm (training stats) + ELU, emit f16 for next GEMM
    fill_f32<<<1, TB, 0, stream>>>(stats, 0.f, 2 * HIDD);
    bn_stats<<<1024, TB, 0, stream>>>(agg, stats);
    bn_elu_f16<<<gridElem, TB, 0, stream>>>(agg, stats, gamma, beta, hin16);
  }

  // head: out = h2 @ Wout + bout  (Wout^T padded 40 -> 48 rows for WMMA)
  convert_transpose_f16<<<64, TB, 0, stream>>>(Wout, wt16, NCLS, NCLS_PAD);
  wmma_gemm_f16f32<<<gemmGridO, 32, 0, stream>>>(hin16, wt16, h, NCLS_PAD);
  out_bias<<<gridElem, TB, 0, stream>>>(h, bout, out);
}
